// GraphAwareContextEncoder_23467701305781
// MI455X (gfx1250) — matrix operand order repaired and checked by reading, not compiled, not run on hardware
//
#include <hip/hip_runtime.h>
#include <math.h>

constexpr int BB  = 4;
constexpr int LL  = 2048;
constexpr int DDim= 128;
constexpr int HH  = 4;
constexpr int DHH = 32;
constexpr int FFD = 256;
constexpr int NLAY= 2;
constexpr int MM  = BB * LL;          // 8192 tokens
constexpr int MW  = LL / 32;          // 64 mask words per row

typedef __attribute__((ext_vector_type(16))) __bf16       v16bf;
typedef __attribute__((ext_vector_type(8)))  float        v8f;
typedef __attribute__((ext_vector_type(8)))  unsigned int v8u;

union Frag16 { v8u u; v16bf bf; };

__device__ __forceinline__ unsigned short f2bf(float f) {
  unsigned int u = __builtin_bit_cast(unsigned int, f);
  u += 0x7fffu + ((u >> 16) & 1u);              // round-to-nearest-even
  return (unsigned short)(u >> 16);
}

__device__ __forceinline__ v8f wmma_bf16(const Frag16& a, const Frag16& b, v8f c) {
  return __builtin_amdgcn_wmma_f32_16x16x32_bf16(false, a.bf, false, b.bf,
                                                 (short)0, c, false, false);
}

// ---------------- weight fp32[K,N] -> bf16[N,K] (transposed) ----------------
__global__ __launch_bounds__(256)
void w2bf_t_kernel(const float* __restrict__ W, unsigned short* __restrict__ out,
                   int K, int N) {
  int i = blockIdx.x * 256 + threadIdx.x;
  if (i >= K * N) return;
  int k = i / N, n = i % N;
  out[(size_t)n * K + k] = f2bf(W[i]);
}

// --- mask fp32 -> bit-packed (1 = masked); need_fix diag folded; full flag --
__global__ __launch_bounds__(256)
void maskpack_kernel(const float* __restrict__ mask, unsigned int* __restrict__ bits,
                     int* __restrict__ fullrow) {
  int wv = threadIdx.x >> 5, lane = threadIdx.x & 31;
  int row = blockIdx.x * 8 + wv;                       // row in [0, B*L)
  const float* mr = mask + (size_t)row * LL;
  unsigned int any = 0u, all = 0xffffffffu;
  for (int w = 0; w < MW; ++w) {
    bool pred = mr[w * 32 + lane] > 0.5f;
    unsigned int bw = (unsigned int)__ballot(pred);    // wave32: low 32 bits
    if (lane == 0) bits[(size_t)row * MW + w] = bw;
    any |= bw; all &= bw;
  }
  if (lane == 0) {
    fullrow[row] = (all == 0xffffffffu) ? 1 : 0;       // torch fully-masked fixup
    if (any == 0u) {                                   // torch need_fix quirk
      int qg = row & (LL - 1);
      bits[(size_t)row * MW + (qg >> 5)] |= 1u << (qg & 31);
    }
  }
}

// ---------------- fused input projections + pre-LN -> x fp32 ----------------
__global__ __launch_bounds__(256)
void fuse_input_kernel(const float* __restrict__ tf, const float* __restrict__ pf,
                       const float* __restrict__ df,
                       const float* __restrict__ tW, const float* __restrict__ tb,
                       const float* __restrict__ pW, const float* __restrict__ pb,
                       const float* __restrict__ dW, const float* __restrict__ db,
                       const float* __restrict__ g,  const float* __restrict__ bt,
                       float* __restrict__ x) {
  int wv = threadIdx.x >> 5, lane = threadIdx.x & 31;
  int tok = blockIdx.x * 8 + wv;
  const float* t = tf + (size_t)tok * 15;
  const float* p = pf + (size_t)tok * 6;
  const float* d = df + (size_t)tok * 4;
  float v[4]; float s = 0.f;
#pragma unroll
  for (int j = 0; j < 4; ++j) {
    int c = lane + 32 * j;
    float acc = tb[c] + pb[c] + db[c];
#pragma unroll
    for (int i = 0; i < 15; ++i) acc += t[i] * tW[i * DDim + c];
#pragma unroll
    for (int i = 0; i < 6;  ++i) acc += p[i] * pW[i * DDim + c];
#pragma unroll
    for (int i = 0; i < 4;  ++i) acc += d[i] * dW[i * DDim + c];
    v[j] = acc; s += acc;
  }
  s += __shfl_xor(s,1); s += __shfl_xor(s,2); s += __shfl_xor(s,4);
  s += __shfl_xor(s,8); s += __shfl_xor(s,16);
  float mu = s * (1.0f / DDim);
  float s2 = 0.f;
#pragma unroll
  for (int j = 0; j < 4; ++j) { float dd = v[j] - mu; s2 += dd * dd; }
  s2 += __shfl_xor(s2,1); s2 += __shfl_xor(s2,2); s2 += __shfl_xor(s2,4);
  s2 += __shfl_xor(s2,8); s2 += __shfl_xor(s2,16);
  float rn = rsqrtf(s2 * (1.0f / DDim) + 1e-5f);
#pragma unroll
  for (int j = 0; j < 4; ++j) {
    int c = lane + 32 * j;
    x[(size_t)tok * DDim + c] = (v[j] - mu) * rn * g[c] + bt[c];
  }
}

// ---------------- LayerNorm: fp32 in -> bf16 out ----------------
__global__ __launch_bounds__(256)
void ln_bf16_kernel(const float* __restrict__ x, const float* __restrict__ g,
                    const float* __restrict__ bt, unsigned short* __restrict__ out) {
  int wv = threadIdx.x >> 5, lane = threadIdx.x & 31;
  int tok = blockIdx.x * 8 + wv;
  const float* row = x + (size_t)tok * DDim;
  float v[4]; float s = 0.f;
#pragma unroll
  for (int j = 0; j < 4; ++j) { v[j] = row[lane + 32 * j]; s += v[j]; }
  s += __shfl_xor(s,1); s += __shfl_xor(s,2); s += __shfl_xor(s,4);
  s += __shfl_xor(s,8); s += __shfl_xor(s,16);
  float mu = s * (1.0f / DDim);
  float s2 = 0.f;
#pragma unroll
  for (int j = 0; j < 4; ++j) { float d = v[j] - mu; s2 += d * d; }
  s2 += __shfl_xor(s2,1); s2 += __shfl_xor(s2,2); s2 += __shfl_xor(s2,4);
  s2 += __shfl_xor(s2,8); s2 += __shfl_xor(s2,16);
  float rn = rsqrtf(s2 * (1.0f / DDim) + 1e-5f);
#pragma unroll
  for (int j = 0; j < 4; ++j) {
    int c = lane + 32 * j;
    out[(size_t)tok * DDim + c] = f2bf((v[j] - mu) * rn * g[c] + bt[c]);
  }
}

// ---------------- LayerNorm: fp32 in -> fp32 out (post-LN) ----------------
__global__ __launch_bounds__(256)
void ln_f32_kernel(const float* __restrict__ x, const float* __restrict__ g,
                   const float* __restrict__ bt, float* __restrict__ out) {
  int wv = threadIdx.x >> 5, lane = threadIdx.x & 31;
  int tok = blockIdx.x * 8 + wv;
  const float* row = x + (size_t)tok * DDim;
  float v[4]; float s = 0.f;
#pragma unroll
  for (int j = 0; j < 4; ++j) { v[j] = row[lane + 32 * j]; s += v[j]; }
  s += __shfl_xor(s,1); s += __shfl_xor(s,2); s += __shfl_xor(s,4);
  s += __shfl_xor(s,8); s += __shfl_xor(s,16);
  float mu = s * (1.0f / DDim);
  float s2 = 0.f;
#pragma unroll
  for (int j = 0; j < 4; ++j) { float d = v[j] - mu; s2 += d * d; }
  s2 += __shfl_xor(s2,1); s2 += __shfl_xor(s2,2); s2 += __shfl_xor(s2,4);
  s2 += __shfl_xor(s2,8); s2 += __shfl_xor(s2,16);
  float rn = rsqrtf(s2 * (1.0f / DDim) + 1e-5f);
#pragma unroll
  for (int j = 0; j < 4; ++j) {
    int c = lane + 32 * j;
    out[(size_t)tok * DDim + c] = (v[j] - mu) * rn * g[c] + bt[c];
  }
}

// -------- WMMA GEMM: 16x32 tile per wave, A fragment reused across N --------
// EPI: 0=store bf16   1=store bf16 + V-transpose scatter
//      2=residual add into fp32    3=GELU then store bf16
template<int EPI>
__global__ __launch_bounds__(256)
void gemm_bf16_kernel(const unsigned short* __restrict__ A,
                      const unsigned short* __restrict__ Bt,
                      const float* __restrict__ bias,
                      int Mtiles, int Ntiles, int Kdim,
                      unsigned short* __restrict__ outb,
                      float* __restrict__ resid,
                      unsigned short* __restrict__ vt) {
  int wv = threadIdx.x >> 5, lane = threadIdx.x & 31;
  int pairs = Ntiles >> 1;
  int gw = blockIdx.x * 8 + wv;
  if (gw >= Mtiles * pairs) return;
  int mt = gw / pairs, np = gw % pairs;
  int lo = lane & 15, hi = lane >> 4;
  const unsigned short* arow  = A  + (size_t)(mt * 16 + lo) * Kdim;
  const unsigned short* brow0 = Bt + (size_t)(np * 32 + lo) * Kdim;
  const unsigned short* brow1 = brow0 + (size_t)16 * Kdim;
  v8f acc0 = {}, acc1 = {};
  for (int k0 = 0; k0 < Kdim; k0 += 32) {
    Frag16 af, b0, b1;
#pragma unroll
    for (int j = 0; j < 8; ++j) {
      af.u[j] = *(const unsigned int*)(arow  + (k0 + 2*j + 16*hi));
      b0.u[j] = *(const unsigned int*)(brow0 + (k0 + 2*j + 16*hi));
      b1.u[j] = *(const unsigned int*)(brow1 + (k0 + 2*j + 16*hi));
    }
    acc0 = wmma_bf16(af, b0, acc0);
    acc1 = wmma_bf16(af, b1, acc1);
  }
  int Nfull = Ntiles * 16;
#pragma unroll
  for (int half = 0; half < 2; ++half) {
    int n = np * 32 + half * 16 + lo;
    float bval = bias[n];
    v8f& acc = half ? acc1 : acc0;
#pragma unroll
    for (int r = 0; r < 8; ++r) {
      int m = mt * 16 + r + 8 * hi;
      float v = acc[r] + bval;
      if (EPI == 0 || EPI == 1) {
        outb[(size_t)m * Nfull + n] = f2bf(v);
        if (EPI == 1) {
          int bidx = m >> 11, tok = m & (LL - 1);
          int h = n >> 5, dh = n & 31;
          vt[((size_t)((bidx * HH + h) * DHH + dh)) * LL + tok] = f2bf(v);
        }
      } else if (EPI == 2) {
        size_t i = (size_t)m * DDim + n;
        resid[i] = resid[i] + v;
      } else {
        float gl = 0.5f * v * (1.0f + erff(v * 0.70710678118654752f));
        outb[(size_t)m * Nfull + n] = f2bf(gl);
      }
    }
  }
}

// ------ Flash attention (max-free softmax): one wave per (b,h,q-tile) -------
// Scores are bounded by construction; exp is clamped at 80 so sums stay finite.
// Deferred row-sum: per-lane partial sums, one butterfly after the k-loop.
__global__ __launch_bounds__(256)
void attn_kernel(const unsigned short* __restrict__ Q,
                 const unsigned short* __restrict__ Kc,
                 const unsigned short* __restrict__ Vt,
                 const unsigned int* __restrict__ mbits,
                 const int* __restrict__ fullrow,
                 unsigned short* __restrict__ outb) {
  __shared__ __align__(16) unsigned short P[8][16 * 32];
  int wv = threadIdx.x >> 5, lane = threadIdx.x & 31;
  int gw = blockIdx.x * 8 + wv;
  int qt = gw & (LL / 16 - 1);
  int h  = (gw >> 7) & (HH - 1);
  int b  = gw >> 9;
  int lo = lane & 15, hi = lane >> 4;
  const float scale = 0.17677669529663689f;   // 1/sqrt(32)

  Frag16 qf;
  {
    const unsigned short* qr = Q + (size_t)(b * LL + qt * 16 + lo) * DDim + h * DHH;
#pragma unroll
    for (int j = 0; j < 8; ++j)
      qf.u[j] = *(const unsigned int*)(qr + (2*j + 16*hi));
  }

  float ps[8]; int qg[8], fullq[8];
  const unsigned int* mrowp[8];
#pragma unroll
  for (int r = 0; r < 8; ++r) {
    ps[r] = 0.f;
    qg[r] = qt * 16 + r + 8 * hi;
    mrowp[r] = mbits + (size_t)(b * LL + qg[r]) * MW;
    fullq[r] = fullrow[b * LL + qg[r]];
  }
  v8f o0 = {}, o1 = {};
  v8f zc = {};

  for (int k0 = 0; k0 < LL; k0 += 32) {
    Frag16 kf0, kf1;
    const unsigned short* kr0 = Kc + (size_t)(b * LL + k0 + lo) * DDim + h * DHH;
    const unsigned short* kr1 = kr0 + (size_t)16 * DDim;
    __builtin_prefetch((const void*)(kr0 + (size_t)32 * DDim), 0, 1);
#pragma unroll
    for (int j = 0; j < 8; ++j) {
      kf0.u[j] = *(const unsigned int*)(kr0 + (2*j + 16*hi));
      kf1.u[j] = *(const unsigned int*)(kr1 + (2*j + 16*hi));
    }
    v8f s0 = wmma_bf16(qf, kf0, zc);
    v8f s1 = wmma_bf16(qf, kf1, zc);

    int kg0 = k0 + lo, kg1 = kg0 + 16;
    int widx = k0 >> 5;
#pragma unroll
    for (int r = 0; r < 8; ++r) {
      unsigned int wbits = mrowp[r][widx];          // broadcast within half-wave
      bool mk0 = (wbits >> lo) & 1u;
      bool mk1 = (wbits >> (lo + 16)) & 1u;
      bool fd0 = mk0 && fullq[r] && (qg[r] == kg0); // torch: score := 0 on diag
      bool fd1 = mk1 && fullq[r] && (qg[r] == kg1);
      float sv0 = fd0 ? 0.f : (s0[r] * scale + (mk0 ? -1e30f : 0.0f));
      float sv1 = fd1 ? 0.f : (s1[r] * scale + (mk1 ? -1e30f : 0.0f));
      float p0 = __expf(fminf(sv0, 80.f));
      float p1 = __expf(fminf(sv1, 80.f));
      ps[r] += p0 + p1;                             // deferred row-sum
      int qrow = r + 8 * hi;
      P[wv][qrow * 32 + lo]      = f2bf(p0);
      P[wv][qrow * 32 + lo + 16] = f2bf(p1);
    }
    asm volatile("s_wait_dscnt 0" ::: "memory");    // same-wave LDS RAW
    Frag16 pfr;
    const unsigned short* pr = &P[wv][lo * 32];
#pragma unroll
    for (int j = 0; j < 8; ++j)
      pfr.u[j] = *(const unsigned int*)(pr + (2*j + 16*hi));
    Frag16 vf0, vf1;
    const unsigned short* vr0 = Vt + (size_t)((b * HH + h) * DHH + lo) * LL + k0;
    const unsigned short* vr1 = vr0 + (size_t)16 * LL;
    __builtin_prefetch((const void*)(vr0 + 32), 0, 1);
#pragma unroll
    for (int j = 0; j < 8; ++j) {
      vf0.u[j] = *(const unsigned int*)(vr0 + (2*j + 16*hi));
      vf1.u[j] = *(const unsigned int*)(vr1 + (2*j + 16*hi));
    }
    o0 = wmma_bf16(pfr, vf0, o0);
    o1 = wmma_bf16(pfr, vf1, o1);
  }

#pragma unroll
  for (int r = 0; r < 8; ++r) {
    float t = ps[r];
    t += __shfl_xor(t, 1); t += __shfl_xor(t, 2);
    t += __shfl_xor(t, 4); t += __shfl_xor(t, 8);
    float inv = 1.0f / t;
    int token = b * LL + qt * 16 + r + 8 * hi;
    outb[(size_t)token * DDim + h * DHH + lo]      = f2bf(o0[r] * inv);
    outb[(size_t)token * DDim + h * DHH + lo + 16] = f2bf(o1[r] * inv);
  }
}

// ---------------- mean over L ----------------
__global__ __launch_bounds__(256)
void mean_kernel(const float* __restrict__ enc, float* __restrict__ gctx) {
  int wv = threadIdx.x >> 5, lane = threadIdx.x & 31;
  int gid = blockIdx.x * 8 + wv;        // 0..511
  int b = gid >> 7, d = gid & 127;
  const float* p = enc + (size_t)b * LL * DDim + d;
  float s = 0.f;
  for (int l = lane; l < LL; l += 32) s += p[(size_t)l * DDim];
  s += __shfl_xor(s,1); s += __shfl_xor(s,2); s += __shfl_xor(s,4);
  s += __shfl_xor(s,8); s += __shfl_xor(s,16);
  if (lane == 0) gctx[b * DDim + d] = s * (1.0f / LL);
}

// ---------------- fused = gctx + 0.1*lg; out = fused @ outW + out_b ---------
__global__ __launch_bounds__(256)
void outproj_kernel(const float* __restrict__ gctx, const float* __restrict__ lg,
                    const float* __restrict__ W, const float* __restrict__ bias,
                    float* __restrict__ out) {
  int tid = blockIdx.x * 256 + threadIdx.x;
  if (tid >= BB * DDim) return;
  int b = tid >> 7, n = tid & 127;
  float acc = bias[n];
  for (int k = 0; k < DDim; ++k)
    acc += (gctx[b * DDim + k] + 0.1f * lg[b * DDim + k]) * W[k * DDim + n];
  out[b * DDim + n] = acc;
}

extern "C" void kernel_launch(void* const* d_in, const int* in_sizes, int n_in,
                              void* d_out, int out_size, void* d_ws, size_t ws_size,
                              hipStream_t stream) {
  const float* lg    = (const float*)d_in[0];
  const float* tfeat = (const float*)d_in[1];
  const float* pfeat = (const float*)d_in[2];
  const float* dfeat = (const float*)d_in[3];
  const float* mask  = (const float*)d_in[4];
  const float* taskW = (const float*)d_in[5];  const float* taskb = (const float*)d_in[6];
  const float* privW = (const float*)d_in[7];  const float* privb = (const float*)d_in[8];
  const float* dagW  = (const float*)d_in[9];  const float* dagb  = (const float*)d_in[10];
  const float* pre_g = (const float*)d_in[11]; const float* pre_b = (const float*)d_in[12];
  const float* post_g= (const float*)d_in[13]; const float* post_b= (const float*)d_in[14];
  const float* outW  = (const float*)d_in[15]; const float* outb_ = (const float*)d_in[16];
  const float* qW = (const float*)d_in[17]; const float* qb_ = (const float*)d_in[18];
  const float* kW = (const float*)d_in[19]; const float* kb_ = (const float*)d_in[20];
  const float* vW = (const float*)d_in[21]; const float* vb_ = (const float*)d_in[22];
  const float* oW = (const float*)d_in[23]; const float* ob_ = (const float*)d_in[24];
  const float* n1g = (const float*)d_in[25]; const float* n1b = (const float*)d_in[26];
  const float* n2g = (const float*)d_in[27]; const float* n2b = (const float*)d_in[28];
  const float* f1W = (const float*)d_in[29]; const float* f1b = (const float*)d_in[30];
  const float* f2W = (const float*)d_in[31]; const float* f2b = (const float*)d_in[32];

  char* w = (char*)d_ws;
  auto take = [&](size_t bytes) -> char* {
    char* p = w; w += (bytes + 255) & ~(size_t)255; return p;
  };
  float*          x    = (float*)take((size_t)MM * DDim * 4);
  unsigned short* xn   = (unsigned short*)take((size_t)MM * DDim * 2);
  unsigned short* qh   = (unsigned short*)take((size_t)MM * DDim * 2);
  unsigned short* kh   = (unsigned short*)take((size_t)MM * DDim * 2);
  unsigned short* vh   = (unsigned short*)take((size_t)MM * DDim * 2);
  unsigned short* vt   = (unsigned short*)take((size_t)MM * DDim * 2);
  unsigned short* at   = (unsigned short*)take((size_t)MM * DDim * 2);
  unsigned short* hb   = (unsigned short*)take((size_t)MM * DDim * 2);
  unsigned short* ffb  = (unsigned short*)take((size_t)MM * FFD * 2);
  float*          enc  = (float*)take((size_t)MM * DDim * 4);
  float*          gctx = (float*)take((size_t)BB * DDim * 4);
  unsigned int*   mbits= (unsigned int*)take((size_t)MM * MW * 4);
  int*            frow = (int*)take((size_t)MM * 4);
  unsigned short* wt   = (unsigned short*)take((size_t)NLAY * 131072 * 2);

  // weight conversion (fp32 [K,N] -> bf16 [N,K])
  for (int i = 0; i < NLAY; ++i) {
    unsigned short* lw = wt + (size_t)i * 131072;
    w2bf_t_kernel<<<64, 256, 0, stream>>>(qW + (size_t)i*DDim*DDim, lw + 0,     DDim, DDim);
    w2bf_t_kernel<<<64, 256, 0, stream>>>(kW + (size_t)i*DDim*DDim, lw + 16384, DDim, DDim);
    w2bf_t_kernel<<<64, 256, 0, stream>>>(vW + (size_t)i*DDim*DDim, lw + 32768, DDim, DDim);
    w2bf_t_kernel<<<64, 256, 0, stream>>>(oW + (size_t)i*DDim*DDim, lw + 49152, DDim, DDim);
    w2bf_t_kernel<<<128,256, 0, stream>>>(f1W + (size_t)i*DDim*FFD, lw + 65536, DDim, FFD);
    w2bf_t_kernel<<<128,256, 0, stream>>>(f2W + (size_t)i*FFD*DDim, lw + 98304, FFD, DDim);
  }
  maskpack_kernel<<<MM / 8, 256, 0, stream>>>(mask, mbits, frow);
  fuse_input_kernel<<<MM / 8, 256, 0, stream>>>(tfeat, pfeat, dfeat,
      taskW, taskb, privW, privb, dagW, dagb, pre_g, pre_b, x);

  for (int i = 0; i < NLAY; ++i) {
    unsigned short* lw = wt + (size_t)i * 131072;
    ln_bf16_kernel<<<MM / 8, 256, 0, stream>>>(x, n1g + i*DDim, n1b + i*DDim, xn);
    gemm_bf16_kernel<0><<<256, 256, 0, stream>>>(xn, lw + 0,     qb_ + i*DDim, MM/16, 8, DDim, qh, nullptr, nullptr);
    gemm_bf16_kernel<0><<<256, 256, 0, stream>>>(xn, lw + 16384, kb_ + i*DDim, MM/16, 8, DDim, kh, nullptr, nullptr);
    gemm_bf16_kernel<1><<<256, 256, 0, stream>>>(xn, lw + 32768, vb_ + i*DDim, MM/16, 8, DDim, vh, nullptr, vt);
    attn_kernel<<<256, 256, 0, stream>>>(qh, kh, vt, mbits, frow, at);
    gemm_bf16_kernel<2><<<256, 256, 0, stream>>>(at, lw + 49152, ob_ + i*DDim, MM/16, 8, DDim, nullptr, x, nullptr);
    ln_bf16_kernel<<<MM / 8, 256, 0, stream>>>(x, n2g + i*DDim, n2b + i*DDim, hb);
    gemm_bf16_kernel<3><<<512, 256, 0, stream>>>(hb, lw + 65536, f1b + i*FFD, MM/16, 16, DDim, ffb, nullptr, nullptr);
    gemm_bf16_kernel<2><<<256, 256, 0, stream>>>(ffb, lw + 98304, f2b + i*DDim, MM/16, 8, FFD, nullptr, x, nullptr);
  }

  ln_f32_kernel<<<MM / 8, 256, 0, stream>>>(x, post_g, post_b, enc);
  mean_kernel<<<64, 256, 0, stream>>>(enc, gctx);
  outproj_kernel<<<2, 256, 0, stream>>>(gctx, lg, outW, outb_, (float*)d_out);
}
